// TransformerEncoderLayerWithAttn_81028853006916
// MI455X (gfx1250) — matrix operand order repaired and checked by reading, not compiled, not run on hardware
//
#include <hip/hip_runtime.h>
#include <hip/hip_bf16.h>

typedef __bf16 bf16;
typedef __attribute__((ext_vector_type(16))) __bf16 v16bf;
typedef __attribute__((ext_vector_type(8)))  float  v8f;

#define S_DIM 1024
#define B_DIM 8
#define D_DIM 1024
#define H_DIM 16
#define HD_DIM 64
#define F_DIM 4096
#define NROWS (S_DIM * B_DIM)   // 8192

static __device__ __forceinline__ v8f wmma_bf16(v16bf a, v16bf b, v8f c) {
  // D = A(16x32 bf16) * B(32x16 bf16) + C(16x16 f32)
  return __builtin_amdgcn_wmma_f32_16x16x32_bf16(false, a, false, b, (short)0, c,
                                                 false, false);
}

// ---------------------------------------------------------------- conversions
__global__ void cvt_f32_to_bf16(const float* __restrict__ in,
                                bf16* __restrict__ out, int n) {
  int i = blockIdx.x * blockDim.x + threadIdx.x;
  int stride = gridDim.x * blockDim.x;
  for (; i < n; i += stride) out[i] = (bf16)in[i];
}

// ---------------------------------------------------------------- GEMM
// C[N,M] = X[N,K](bf16,row) * W[M,K](bf16,row)^T + bias, fused epilogue.
// Tiles staged with GLOBAL_LOAD_ASYNC_TO_LDS_B128 (ASYNCcnt-tracked DMA):
// no VGPR round-trip, overlapped with WMMA compute via double buffering.
enum { MODE_QKV = 0, MODE_F32_RES = 1, MODE_BF16_RELU = 2 };

template <int MODE>
__global__ __launch_bounds__(256) void gemm_wmma(
    const bf16* __restrict__ X, const bf16* __restrict__ W,
    const float* __restrict__ bias, void* __restrict__ outp,
    const float* __restrict__ res, int K, int ldout) {
  __shared__ bf16 sA[2][128][32];
  __shared__ bf16 sB[2][128][32];

  const int rowBase = blockIdx.y * 128;
  const int colBase = blockIdx.x * 128;
  const int t = threadIdx.x;
  const int wv = t >> 5;
  const int lane = t & 31;
  const int lm = lane & 15;
  const int lh = lane >> 4;
  const int waveM = wv >> 2;  // 0..1 : 64-row group
  const int waveN = wv & 3;   // 0..3 : 32-col group

  const int lr = t >> 1;        // 0..127 tile row for loads
  const int lc = (t & 1) << 4;  // 0 or 16 (bf16 elems)

  // async DMA of one 128x32 bf16 tile of X and W into LDS buffer `buf`.
  // INST_OFFSET is added to both the global and LDS address, so two b128
  // ops at offsets 0/16 move this thread's 32-byte chunk.
  auto load_tile_async = [&](int kt, int buf) {
    size_t k0 = (size_t)kt * 32;
    unsigned long long ax =
        (unsigned long long)(X + (size_t)(rowBase + lr) * K + k0 + lc);
    unsigned long long aw =
        (unsigned long long)(W + (size_t)(colBase + lr) * K + k0 + lc);
    unsigned la = (unsigned)(unsigned long long)(&sA[buf][lr][lc]);
    unsigned lb = (unsigned)(unsigned long long)(&sB[buf][lr][lc]);
    asm volatile(
        "global_load_async_to_lds_b128 %0, %2, off\n\t"
        "global_load_async_to_lds_b128 %0, %2, off offset:16\n\t"
        "global_load_async_to_lds_b128 %1, %3, off\n\t"
        "global_load_async_to_lds_b128 %1, %3, off offset:16"
        :: "v"(la), "v"(lb), "v"(ax), "v"(aw)
        : "memory");
  };

  v8f zero = {0.f, 0.f, 0.f, 0.f, 0.f, 0.f, 0.f, 0.f};
  v8f acc[4][2];
#pragma unroll
  for (int mi = 0; mi < 4; ++mi)
#pragma unroll
    for (int ni = 0; ni < 2; ++ni) acc[mi][ni] = zero;

  const int NT = K >> 5;
  load_tile_async(0, 0);
  asm volatile("s_wait_asynccnt 0x0" ::: "memory");
  __syncthreads();
  for (int kt = 0; kt < NT; ++kt) {
    int cur = kt & 1;
    if (kt + 1 < NT) load_tile_async(kt + 1, cur ^ 1);
    v16bf af[4], bfg[2];
#pragma unroll
    for (int mi = 0; mi < 4; ++mi) {
      const bf16* pr = &sA[cur][waveM * 64 + mi * 16 + lm][0];
#pragma unroll
      for (int i = 0; i < 16; ++i) {
        int kk = ((i < 8) ? i : (i + 8)) + 8 * lh;  // A-layout K mapping
        af[mi][i] = pr[kk];
      }
    }
#pragma unroll
    for (int ni = 0; ni < 2; ++ni) {
      const bf16* pr = &sB[cur][waveN * 32 + ni * 16 + lm][0];
#pragma unroll
      for (int i = 0; i < 16; ++i) {
        int kk = ((i < 8) ? i : (i + 8)) + 8 * lh;  // B-layout K mapping
        bfg[ni][i] = pr[kk];
      }
    }
#pragma unroll
    for (int mi = 0; mi < 4; ++mi)
#pragma unroll
      for (int ni = 0; ni < 2; ++ni)
        acc[mi][ni] = wmma_bf16(af[mi], bfg[ni], acc[mi][ni]);
    // next tile's DMA must be fully in LDS before anyone reads it
    asm volatile("s_wait_asynccnt 0x0" ::: "memory");
    __syncthreads();
  }

  // epilogue: C layout — row = 8*lh + j, col = lm within each 16x16 tile
#pragma unroll
  for (int mi = 0; mi < 4; ++mi) {
#pragma unroll
    for (int ni = 0; ni < 2; ++ni) {
#pragma unroll
      for (int j = 0; j < 8; ++j) {
        int r = rowBase + waveM * 64 + mi * 16 + lh * 8 + j;
        int c = colBase + waveN * 32 + ni * 16 + lm;
        float val = acc[mi][ni][j] + bias[c];
        if (MODE == MODE_QKV) {
          int sel = c >> 10, rem = c & 1023;
          int h = rem >> 6, d = rem & 63;
          int s = r >> 3, b = r & 7;
          if (sel == 0) val *= 0.125f;  // 1/sqrt(HD)
          bf16* o = (bf16*)outp;
          size_t idx =
              ((((size_t)sel * B_DIM + b) * H_DIM + h) * S_DIM + s) * HD_DIM + d;
          o[idx] = (bf16)val;
        } else if (MODE == MODE_BF16_RELU) {
          bf16* o = (bf16*)outp;
          o[(size_t)r * ldout + c] = (bf16)fmaxf(val, 0.0f);
        } else {  // MODE_F32_RES
          float* o = (float*)outp;
          o[(size_t)r * ldout + c] = val + res[(size_t)r * ldout + c];
        }
      }
    }
  }
}

// ---------------------------------------------------------------- attention
// Flash-style: 4 waves/WG, each wave owns 16 query rows; stream 64-key tiles.
// q,k,v laid out [B*H][S][HD] bf16; q pre-scaled by 1/sqrt(HD).
__global__ __launch_bounds__(128) void attn_flash(const bf16* __restrict__ q,
                                                  const bf16* __restrict__ k,
                                                  const bf16* __restrict__ v,
                                                  bf16* __restrict__ ctx) {
  __shared__ bf16 vT[64][72];       // V^T tile: [d][key], padded rows
  __shared__ bf16 pT[4][16][64];    // per-wave P bounce: [qrow][key]

  const int bh = blockIdx.y;
  const int b = bh / H_DIM, h = bh % H_DIM;
  const size_t base = (size_t)bh * S_DIM * HD_DIM;
  const int t = threadIdx.x;
  const int wv = t >> 5, lane = t & 31;
  const int lm = lane & 15, lh = lane >> 4;
  const int q0 = blockIdx.x * 64 + wv * 16;

  // Q fragments (A-layout), loaded once: 16 rows x 64 dims = 2 k-steps
  v16bf aq[2];
#pragma unroll
  for (int kk = 0; kk < 2; ++kk)
#pragma unroll
    for (int i = 0; i < 16; ++i) {
      int d = kk * 32 + ((i < 8) ? i : (i + 8)) + 8 * lh;
      aq[kk][i] = q[base + (size_t)(q0 + lm) * HD_DIM + d];
    }

  v8f zero = {0.f, 0.f, 0.f, 0.f, 0.f, 0.f, 0.f, 0.f};
  float m_i[8], l_i[8];
  v8f acc[4];
#pragma unroll
  for (int j = 0; j < 8; ++j) { m_i[j] = -1e30f; l_i[j] = 0.f; }
#pragma unroll
  for (int nt = 0; nt < 4; ++nt) acc[nt] = zero;

  for (int jt = 0; jt < S_DIM / 64; ++jt) {
    const int kbase = jt * 64;
    __syncthreads();
    // stage V tile transposed into LDS (coalesced global reads)
    for (int idx = t; idx < 64 * 64; idx += 128) {
      int key = idx >> 6, d = idx & 63;
      vT[d][key] = v[base + (size_t)(kbase + key) * HD_DIM + d];
    }
    __syncthreads();

    // S = Q * K^T : B-fragments straight from global (rows of K contiguous)
    v8f sacc[4];
#pragma unroll
    for (int nt = 0; nt < 4; ++nt) {
      v8f c = zero;
#pragma unroll
      for (int kk = 0; kk < 2; ++kk) {
        v16bf bk;
        int key = kbase + nt * 16 + lm;
        const bf16* kp = &k[base + (size_t)key * HD_DIM + kk * 32 + lh * 8];
#pragma unroll
        for (int i = 0; i < 16; ++i) bk[i] = kp[(i < 8) ? i : (i + 8)];
        c = wmma_bf16(aq[kk], bk, c);
      }
      sacc[nt] = c;
    }

    // online softmax; row r = 8*lh + j lives in VGPR j across the 16-lane half
    float rs[8];
#pragma unroll
    for (int j = 0; j < 8; ++j) {
      float mx = fmaxf(fmaxf(sacc[0][j], sacc[1][j]),
                       fmaxf(sacc[2][j], sacc[3][j]));
#pragma unroll
      for (int off = 1; off < 16; off <<= 1)
        mx = fmaxf(mx, __shfl_xor(mx, off, 32));
      float mnew = fmaxf(m_i[j], mx);
      float alpha = __expf(m_i[j] - mnew);
      m_i[j] = mnew;
      l_i[j] *= alpha;
#pragma unroll
      for (int nt = 0; nt < 4; ++nt) acc[nt][j] *= alpha;
      float sum = 0.f;
#pragma unroll
      for (int nt = 0; nt < 4; ++nt) {
        float p = __expf(sacc[nt][j] - mnew);
        sum += p;
        pT[wv][lh * 8 + j][nt * 16 + lm] = (bf16)p;  // C-layout -> LDS
      }
      rs[j] = sum;
    }
#pragma unroll
    for (int j = 0; j < 8; ++j) {
      float sum = rs[j];
#pragma unroll
      for (int off = 1; off < 16; off <<= 1) sum += __shfl_xor(sum, off, 32);
      l_i[j] += sum;
    }

    // O += P * V : A-fragments of P from LDS, B-fragments of V^T from LDS
    v16bf ap[2];
#pragma unroll
    for (int kk = 0; kk < 2; ++kk)
#pragma unroll
      for (int i = 0; i < 16; ++i) {
        int kkey = kk * 32 + ((i < 8) ? i : (i + 8)) + 8 * lh;
        ap[kk][i] = pT[wv][lm][kkey];
      }
#pragma unroll
    for (int nt = 0; nt < 4; ++nt)
#pragma unroll
      for (int kk = 0; kk < 2; ++kk) {
        v16bf bv;
        const bf16* vp = &vT[nt * 16 + lm][kk * 32 + lh * 8];
#pragma unroll
        for (int i = 0; i < 16; ++i) bv[i] = vp[(i < 8) ? i : (i + 8)];
        acc[nt] = wmma_bf16(ap[kk], bv, acc[nt]);
      }
  }

  // normalize and store ctx as [S][B][D] bf16
#pragma unroll
  for (int j = 0; j < 8; ++j) {
    float inv = 1.0f / l_i[j];
    int s = q0 + lh * 8 + j;
#pragma unroll
    for (int nt = 0; nt < 4; ++nt) {
      int dd = nt * 16 + lm;
      ctx[((size_t)s * B_DIM + b) * D_DIM + h * HD_DIM + dd] =
          (bf16)(acc[nt][j] * inv);
    }
  }
}

// ---------------------------------------------------------------- layernorm
// One row per block (D=1024, 256 threads x 4). In-place capable; optional
// bf16 shadow copy.
__global__ __launch_bounds__(256) void layernorm_kernel(
    float* __restrict__ x, bf16* __restrict__ xbf, const float* __restrict__ g,
    const float* __restrict__ be) {
  __shared__ float red[8];
  __shared__ float red2[8];
  const int row = blockIdx.x;
  const int t = threadIdx.x;
  const int lane = t & 31, wv = t >> 5;
  const size_t basei = (size_t)row * D_DIM;
  float vals[4];
  float s = 0.f;
#pragma unroll
  for (int i = 0; i < 4; ++i) {
    vals[i] = x[basei + t + i * 256];
    s += vals[i];
  }
#pragma unroll
  for (int off = 16; off >= 1; off >>= 1) s += __shfl_xor(s, off, 32);
  if (lane == 0) red[wv] = s;
  __syncthreads();
  float tot = 0.f;
#pragma unroll
  for (int i = 0; i < 8; ++i) tot += red[i];
  float mean = tot * (1.0f / D_DIM);
  float vs = 0.f;
#pragma unroll
  for (int i = 0; i < 4; ++i) {
    float d = vals[i] - mean;
    vs += d * d;
  }
#pragma unroll
  for (int off = 16; off >= 1; off >>= 1) vs += __shfl_xor(vs, off, 32);
  if (lane == 0) red2[wv] = vs;
  __syncthreads();
  float vtot = 0.f;
#pragma unroll
  for (int i = 0; i < 8; ++i) vtot += red2[i];
  float rstd = rsqrtf(vtot * (1.0f / D_DIM) + 1e-5f);
#pragma unroll
  for (int i = 0; i < 4; ++i) {
    int c = t + i * 256;
    float o = (vals[i] - mean) * rstd * g[c] + be[c];
    x[basei + c] = o;
    if (xbf) xbf[basei + c] = (bf16)o;
  }
}

// ---------------------------------------------------------------- launch
extern "C" void kernel_launch(void* const* d_in, const int* in_sizes, int n_in,
                              void* d_out, int out_size, void* d_ws,
                              size_t ws_size, hipStream_t stream) {
  (void)in_sizes; (void)n_in; (void)out_size; (void)ws_size;
  const float* src       = (const float*)d_in[0];
  const float* in_proj_w = (const float*)d_in[1];
  const float* in_proj_b = (const float*)d_in[2];
  const float* out_w     = (const float*)d_in[3];
  const float* out_b     = (const float*)d_in[4];
  const float* w1        = (const float*)d_in[5];
  const float* b1        = (const float*)d_in[6];
  const float* w2        = (const float*)d_in[7];
  const float* b2        = (const float*)d_in[8];
  const float* g1        = (const float*)d_in[9];
  const float* be1       = (const float*)d_in[10];
  const float* g2        = (const float*)d_in[11];
  const float* be2       = (const float*)d_in[12];

  char* ws = (char*)d_ws;
  const size_t MB = 1ull << 20;
  // region 0 [0,64MB):   qkv bf16 (48MB), later reused as ffn hidden (64MB)
  // region 1 [64,80MB):  src bf16 (16MB), later reused as ctx bf16
  // region 2 [80,104MB): bf16 weights (6+2+8+8 MB)
  // region 3 [104,136MB): x f32 (32MB)
  // region 4 [136,152MB): x bf16 (16MB)
  bf16*  qkv_bf  = (bf16*)(ws + 0);
  bf16*  h1_bf   = (bf16*)(ws + 0);
  bf16*  src_bf  = (bf16*)(ws + 64 * MB);
  bf16*  ctx_bf  = (bf16*)(ws + 64 * MB);
  bf16*  wqkv_bf = (bf16*)(ws + 80 * MB);
  bf16*  wout_bf = (bf16*)(ws + 86 * MB);
  bf16*  w1_bf   = (bf16*)(ws + 88 * MB);
  bf16*  w2_bf   = (bf16*)(ws + 96 * MB);
  float* x_f32   = (float*)(ws + 104 * MB);
  bf16*  x_bf    = (bf16*)(ws + 136 * MB);

  cvt_f32_to_bf16<<<512, 256, 0, stream>>>(src, src_bf, NROWS * D_DIM);
  cvt_f32_to_bf16<<<512, 256, 0, stream>>>(in_proj_w, wqkv_bf, 3 * D_DIM * D_DIM);
  cvt_f32_to_bf16<<<256, 256, 0, stream>>>(out_w, wout_bf, D_DIM * D_DIM);
  cvt_f32_to_bf16<<<512, 256, 0, stream>>>(w1, w1_bf, F_DIM * D_DIM);
  cvt_f32_to_bf16<<<512, 256, 0, stream>>>(w2, w2_bf, D_DIM * F_DIM);

  // QKV projection with fused scatter to [3][B][H][S][HD] and q-scale
  gemm_wmma<MODE_QKV><<<dim3(24, 64), 256, 0, stream>>>(
      src_bf, wqkv_bf, in_proj_b, (void*)qkv_bf, nullptr, D_DIM, 0);

  const bf16* qp = qkv_bf;
  const bf16* kp = qkv_bf + (size_t)B_DIM * H_DIM * S_DIM * HD_DIM;
  const bf16* vp = qkv_bf + 2 * (size_t)B_DIM * H_DIM * S_DIM * HD_DIM;
  attn_flash<<<dim3(S_DIM / 64, B_DIM * H_DIM), 128, 0, stream>>>(qp, kp, vp,
                                                                  ctx_bf);

  // out-proj + residual(src) -> y f32
  gemm_wmma<MODE_F32_RES><<<dim3(8, 64), 256, 0, stream>>>(
      ctx_bf, wout_bf, out_b, (void*)x_f32, src, D_DIM, D_DIM);

  // LN1: in-place f32 + bf16 shadow
  layernorm_kernel<<<NROWS, 256, 0, stream>>>(x_f32, x_bf, g1, be1);

  // FFN1: relu(x @ w1^T + b1) -> bf16
  gemm_wmma<MODE_BF16_RELU><<<dim3(32, 64), 256, 0, stream>>>(
      x_bf, w1_bf, b1, (void*)h1_bf, nullptr, D_DIM, F_DIM);

  // FFN2 + residual(x) -> d_out f32
  gemm_wmma<MODE_F32_RES><<<dim3(8, 64), 256, 0, stream>>>(
      h1_bf, w2_bf, b2, d_out, x_f32, F_DIM, D_DIM);

  // LN2 in-place on d_out
  layernorm_kernel<<<NROWS, 256, 0, stream>>>((float*)d_out, nullptr, g2, be2);
}
